// TransformerVocos_15745350107181
// MI455X (gfx1250) — matrix operand order repaired and checked by reading, not compiled, not run on hardware
//
#include <hip/hip_runtime.h>
#include <hip/hip_bf16.h>

// ---------------- types ----------------
typedef __bf16 bf16;
typedef __attribute__((ext_vector_type(8)))  bf16  bf16x8;
typedef __attribute__((ext_vector_type(16))) bf16  bf16x16;
typedef __attribute__((ext_vector_type(8)))  float floatx8;

union BF16x16 { bf16x16 v; bf16x8 h[2]; };

constexpr int kE = 1024;   // embed dim
constexpr int kNH = 16;    // heads
constexpr int kDh = 64;    // head dim

#define WMMA_BF16(A, B, C) \
  __builtin_amdgcn_wmma_f32_16x16x32_bf16(false, (A), false, (B), (short)0, (C), false, false)

// ---------------- f32 -> bf16 convert ----------------
__global__ __launch_bounds__(256) void cvt_f32_bf16(const float* __restrict__ in,
                                                    bf16* __restrict__ out, long n) {
  long i = (long)blockIdx.x * 256 + threadIdx.x;
  if (i < n) out[i] = (bf16)in[i];
}

// ---------------- GEMM: out[M,E] = A[M,E](bf16) @ W[E,E]^T (+bias) ----------------
// Per-wave tile: 32 rows x 64 cols -> 8 WMMAs per 32-wide k-step, A-frags reused
// across 4 B-frags (21 FLOP/byte from cache vs 10.7 for a 16x32 strip).
// A-frag (16x32, contiguous k order matched by B): lane = row, k span hi*16 -> 16 contiguous.
// B-frag (32x16 column-major = rows of W): lane -> col n=lo, k span hi*16 -> 16
// contiguous bf16 from one W row.
template <typename OutT>
__global__ __launch_bounds__(256)
void gemm_xwt(const bf16* __restrict__ A, const bf16* __restrict__ W,
              const float* __restrict__ bias, OutT* __restrict__ out) {
  const int lane = threadIdx.x & 31;
  const int wave = threadIdx.x >> 5;
  const int hi = lane >> 4, lo = lane & 15;
  const int m0 = blockIdx.x * 64 + (wave & 1) * 32;       // 2 m-waves * 32 rows
  const int n0 = blockIdx.y * 256 + (wave >> 1) * 64;     // 4 n-waves * 64 cols

  floatx8 acc[2][4];
#pragma unroll
  for (int mi = 0; mi < 2; ++mi)
#pragma unroll
    for (int j = 0; j < 4; ++j) acc[mi][j] = (floatx8){0,0,0,0,0,0,0,0};

  const bf16* arow0 = A + (size_t)(m0 + lo) * kE + hi * 16;
  const bf16* arow1 = arow0 + (size_t)16 * kE;
  const bf16* brow[4];
#pragma unroll
  for (int j = 0; j < 4; ++j)
    brow[j] = W + (size_t)(n0 + j * 16 + lo) * kE + hi * 16;

  for (int k0 = 0; k0 < kE; k0 += 32) {
    __builtin_prefetch(brow[0] + k0 + 128, 0, 1);
    BF16x16 af0, af1;
    af0.h[0] = *(const bf16x8*)(arow0 + k0);
    af0.h[1] = *(const bf16x8*)(arow0 + k0 + 8);
    af1.h[0] = *(const bf16x8*)(arow1 + k0);
    af1.h[1] = *(const bf16x8*)(arow1 + k0 + 8);
#pragma unroll
    for (int j = 0; j < 4; ++j) {
      bf16x16 bj = *(const bf16x16*)(brow[j] + k0);
      acc[0][j] = WMMA_BF16(af0.v, bj, acc[0][j]);
      acc[1][j] = WMMA_BF16(af1.v, bj, acc[1][j]);
    }
  }

  float bv[4] = {0.f, 0.f, 0.f, 0.f};
  if (bias) {
#pragma unroll
    for (int j = 0; j < 4; ++j) bv[j] = bias[n0 + j * 16 + lo];
  }

#pragma unroll
  for (int mi = 0; mi < 2; ++mi)
#pragma unroll
    for (int r = 0; r < 8; ++r) {
      size_t row = (size_t)(m0 + mi * 16 + r + 8 * hi);
#pragma unroll
      for (int j = 0; j < 4; ++j) {
        float v = acc[mi][j][r] + bv[j];
        if constexpr (sizeof(OutT) == 4) {
          ((float*)out)[row * kE + n0 + j * 16 + lo] = v;
        } else {
          ((bf16*)out)[row * kE + n0 + j * 16 + lo] = (bf16)v;
        }
      }
    }
}

// ---------------- RoPE (in-place, f32 math) + V transpose ----------------
__global__ __launch_bounds__(256)
void rope_vt(bf16* __restrict__ Qb, bf16* __restrict__ Kb,
             const bf16* __restrict__ Vb, bf16* __restrict__ Vt,
             const float* __restrict__ cosT, const float* __restrict__ sinT,
             long n, int Tp) {
  long i = (long)blockIdx.x * 256 + threadIdx.x;
  if (i >= n) return;                 // n = T * NH * 32
  int d = (int)(i & 31);
  int h = (int)((i >> 5) & 15);
  long t = i >> 9;
  int e1 = h * kDh + d, e2 = e1 + 32;
  float c1 = cosT[t * kDh + d],      s1 = sinT[t * kDh + d];
  float c2 = cosT[t * kDh + d + 32], s2 = sinT[t * kDh + d + 32];
  size_t p1 = (size_t)t * kE + e1, p2 = (size_t)t * kE + e2;
  float q1 = (float)Qb[p1], q2 = (float)Qb[p2];
  Qb[p1] = (bf16)(q1 * c1 - q2 * s1);
  Qb[p2] = (bf16)(q2 * c2 + q1 * s2);
  float k1 = (float)Kb[p1], k2 = (float)Kb[p2];
  Kb[p1] = (bf16)(k1 * c1 - k2 * s1);
  Kb[p2] = (bf16)(k2 * c2 + k1 * s2);
  Vt[(size_t)e1 * Tp + t] = Vb[p1];
  Vt[(size_t)e2 * Tp + t] = Vb[p2];
}

// ---------------- flash attention: one wave per (16-query tile, head) ----------------
__device__ inline float red_max16(float x) {
  x = fmaxf(x, __shfl_xor(x, 1));  x = fmaxf(x, __shfl_xor(x, 2));
  x = fmaxf(x, __shfl_xor(x, 4));  x = fmaxf(x, __shfl_xor(x, 8));
  return x;
}
__device__ inline float red_sum16(float x) {
  x += __shfl_xor(x, 1);  x += __shfl_xor(x, 2);
  x += __shfl_xor(x, 4);  x += __shfl_xor(x, 8);
  return x;
}

__global__ __launch_bounds__(256)
void attn_fwd(const bf16* __restrict__ Qb, const bf16* __restrict__ Kb,
              const bf16* __restrict__ Vt, const int* __restrict__ seq,
              bf16* __restrict__ Ab, int T, int Tp, int nseq) {
  __shared__ bf16 psh[8][16 * 32];      // per-wave P staging for C->A relayout
  const int lane = threadIdx.x & 31;
  const int wave = threadIdx.x >> 5;
  const int hi = lane >> 4, lo = lane & 15;

  int job = blockIdx.x * 8 + wave;
  int nTiles = T >> 4;
  if (job >= nTiles * kNH) return;
  int tile = job / kNH, head = job % kNH;
  int t0 = tile * 16;
  int hbase = head * kDh;

  // locate sequence containing this tile (lengths are multiples of 16)
  int s0 = 0, cum = 0;
  for (int b = 0; b < nseq; ++b) {
    int Lb = seq[b];
    if (t0 < cum + Lb) { s0 = cum; break; }
    cum += Lb;
  }
  int ql0 = t0 - s0;

  // Q fragments for d0 = 0 and d0 = 32
  BF16x16 qf0, qf1;
  {
    const bf16* qr = Qb + (size_t)(t0 + lo) * kE + hbase + hi * 16;
    qf0.h[0] = *(const bf16x8*)(qr);       qf0.h[1] = *(const bf16x8*)(qr + 8);
    qf1.h[0] = *(const bf16x8*)(qr + 32);  qf1.h[1] = *(const bf16x8*)(qr + 40);
  }

  floatx8 o0 = {0,0,0,0,0,0,0,0}, o1 = o0, o2 = o0, o3 = o0;
  float m_run[8], l_run[8];
#pragma unroll
  for (int r = 0; r < 8; ++r) { m_run[r] = -1e30f; l_run[r] = 0.f; }

  bf16* pw = &psh[wave][0];

  for (int kb = 0; kb <= ql0 + 15; kb += 32) {
    // ---- S = Q @ K^T for 32 keys (two 16x16 tiles) ----
    floatx8 sc[2];
#pragma unroll
    for (int c = 0; c < 2; ++c) {
      const bf16* kr = Kb + (size_t)(s0 + kb + c * 16 + lo) * kE + hbase + hi * 16;
      bf16x16 kA = *(const bf16x16*)(kr);        // d = 0..31 slice
      bf16x16 kB = *(const bf16x16*)(kr + 32);   // d = 32..63 slice
      floatx8 s = {0,0,0,0,0,0,0,0};
      s = WMMA_BF16(qf0.v, kA, s);
      s = WMMA_BF16(qf1.v, kB, s);
      sc[c] = s;
    }
    // ---- causal mask + scale (1/sqrt(64)) ----
#pragma unroll
    for (int c = 0; c < 2; ++c) {
      int kp = kb + c * 16 + lo;
#pragma unroll
      for (int r = 0; r < 8; ++r) {
        int qpos = ql0 + r + 8 * hi;
        sc[c][r] = (kp <= qpos) ? sc[c][r] * 0.125f : -1e30f;
      }
    }
    // ---- online softmax stats ----
    float alpha[8];
#pragma unroll
    for (int r = 0; r < 8; ++r) {
      float mx = red_max16(fmaxf(sc[0][r], sc[1][r]));
      float mnew = fmaxf(m_run[r], mx);
      alpha[r] = __expf(m_run[r] - mnew);
      m_run[r] = mnew;
    }
#pragma unroll
    for (int c = 0; c < 2; ++c)
#pragma unroll
      for (int r = 0; r < 8; ++r) {
        float pv = __expf(sc[c][r] - m_run[r]);
        sc[c][r] = pv;
        pw[(r + 8 * hi) * 32 + c * 16 + lo] = (bf16)pv;   // C-layout -> LDS
      }
#pragma unroll
    for (int r = 0; r < 8; ++r) {
      float rs = red_sum16(sc[0][r] + sc[1][r]);
      l_run[r] = l_run[r] * alpha[r] + rs;
      o0[r] *= alpha[r]; o1[r] *= alpha[r]; o2[r] *= alpha[r]; o3[r] *= alpha[r];
    }
    // ---- P back from LDS in A-layout (same-wave DS ops are in-order) ----
    BF16x16 pf;
    pf.h[0] = *(const bf16x8*)(pw + lo * 32 + hi * 16);
    pf.h[1] = *(const bf16x8*)(pw + lo * 32 + hi * 16 + 8);
    // ---- O += P @ V  (B-frags contiguous thanks to V^T) ----
    const bf16* vbase = Vt + (size_t)(hbase + lo) * Tp + s0 + kb + hi * 16;
    o0 = WMMA_BF16(pf.v, *(const bf16x16*)(vbase),                   o0);
    o1 = WMMA_BF16(pf.v, *(const bf16x16*)(vbase + (size_t)16 * Tp), o1);
    o2 = WMMA_BF16(pf.v, *(const bf16x16*)(vbase + (size_t)32 * Tp), o2);
    o3 = WMMA_BF16(pf.v, *(const bf16x16*)(vbase + (size_t)48 * Tp), o3);
  }

  // ---- normalize + store bf16 for the final projection GEMM ----
#pragma unroll
  for (int r = 0; r < 8; ++r) {
    float inv = 1.0f / l_run[r];
    bf16* dst = Ab + (size_t)(t0 + r + 8 * hi) * kE + hbase + lo;
    dst[0]  = (bf16)(o0[r] * inv);
    dst[16] = (bf16)(o1[r] * inv);
    dst[32] = (bf16)(o2[r] * inv);
    dst[48] = (bf16)(o3[r] * inv);
  }
}

// ---------------- launch ----------------
extern "C" void kernel_launch(void* const* d_in, const int* in_sizes, int n_in,
                              void* d_out, int out_size, void* d_ws, size_t ws_size,
                              hipStream_t stream) {
  const float* hs   = (const float*)d_in[0];
  const int*   seq  = (const int*)d_in[1];
  const float* cosT = (const float*)d_in[2];
  const float* sinT = (const float*)d_in[3];
  const float* Wq   = (const float*)d_in[4];
  const float* bq   = (const float*)d_in[5];
  const float* Wk   = (const float*)d_in[6];
  const float* Wv   = (const float*)d_in[7];
  const float* bv   = (const float*)d_in[8];
  const float* Wo   = (const float*)d_in[9];
  const float* bo   = (const float*)d_in[10];

  const int T  = in_sizes[0] / kE;     // total packed tokens (6144)
  const int nseq = in_sizes[1];
  const int Tp = T + 64;               // padded rows so masked tail reads stay in-bounds

  char* p = (char*)d_ws;
  auto alloc = [&](size_t bytes) -> void* {
    void* r = (void*)p;
    p += (bytes + 255) & ~(size_t)255;
    return r;
  };
  const size_t szTE = (size_t)T * kE, szEE = (size_t)kE * kE;
  bf16* Xb  = (bf16*)alloc(szTE * 2);
  bf16* Wqb = (bf16*)alloc(szEE * 2);
  bf16* Wkb = (bf16*)alloc(szEE * 2);
  bf16* Wvb = (bf16*)alloc(szEE * 2);
  bf16* Wob = (bf16*)alloc(szEE * 2);
  bf16* Qb  = (bf16*)alloc((size_t)Tp * kE * 2);
  bf16* Kb  = (bf16*)alloc((size_t)Tp * kE * 2);
  bf16* Vb  = (bf16*)alloc(szTE * 2);
  bf16* Vt  = (bf16*)alloc((size_t)kE * Tp * 2);
  bf16* Ab  = (bf16*)alloc(szTE * 2);

  auto blocks = [](long n) { return (unsigned)((n + 255) / 256); };

  cvt_f32_bf16<<<blocks(szTE), 256, 0, stream>>>(hs, Xb, (long)szTE);
  cvt_f32_bf16<<<blocks(szEE), 256, 0, stream>>>(Wq, Wqb, (long)szEE);
  cvt_f32_bf16<<<blocks(szEE), 256, 0, stream>>>(Wk, Wkb, (long)szEE);
  cvt_f32_bf16<<<blocks(szEE), 256, 0, stream>>>(Wv, Wvb, (long)szEE);
  cvt_f32_bf16<<<blocks(szEE), 256, 0, stream>>>(Wo, Wob, (long)szEE);

  dim3 gg(T / 64, kE / 256);           // 64x256 block tile, 8 waves of 32x64
  gemm_xwt<bf16><<<gg, 256, 0, stream>>>(Xb, Wqb, bq, Qb);
  gemm_xwt<bf16><<<gg, 256, 0, stream>>>(Xb, Wkb, (const float*)nullptr, Kb);
  gemm_xwt<bf16><<<gg, 256, 0, stream>>>(Xb, Wvb, bv, Vb);

  long nrope = (long)T * kNH * 32;
  rope_vt<<<blocks(nrope), 256, 0, stream>>>(Qb, Kb, Vb, Vt, cosT, sinT, nrope, Tp);

  int jobs = (T / 16) * kNH;
  attn_fwd<<<(jobs + 7) / 8, 256, 0, stream>>>(Qb, Kb, Vt, seq, Ab, T, Tp, nseq);

  gemm_xwt<float><<<gg, 256, 0, stream>>>(Ab, Wob, bo, (float*)d_out);
}
